// MultiHeadAttention_42150809043172
// MI455X (gfx1250) — matrix operand order repaired and checked by reading, not compiled, not run on hardware
//
#include <hip/hip_runtime.h>
#include <hip/hip_bf16.h>

#define T_TOK 4096
#define C_DIM 1024
#define H_HEADS 16
#define D_HEAD 64

typedef __bf16 bf16_t;
typedef __attribute__((ext_vector_type(16))) __bf16 v16bf;
typedef __attribute__((ext_vector_type(8)))  __bf16 v8bf;
typedef __attribute__((ext_vector_type(8)))  float  v8f;

// ---------------------------------------------------------------------------
// WMMA helpers (wave32, 16x16x32 bf16 -> f32)
// ---------------------------------------------------------------------------
__device__ __forceinline__ v8f wmma_bf16(v16bf a, v16bf b, v8f c) {
  // (neg_a, A, neg_b, B, c_mod, C, reuse_a, reuse_b)
  return __builtin_amdgcn_wmma_f32_16x16x32_bf16(false, a, false, b, (short)0, c,
                                                 false, false);
}

__device__ __forceinline__ v16bf cat8(v8bf lo, v8bf hi) {
  return __builtin_shufflevector(lo, hi, 0, 1, 2, 3, 4, 5, 6, 7,
                                 8, 9, 10, 11, 12, 13, 14, 15);
}

// A-matrix fragment, 16x32 bf16, source row-major [16][K...] with row stride ld.
// ISA layout: lane holds row r=lane%16; K = g*8+{0..7} and 16+g*8+{0..7}, g=lane/16.
__device__ __forceinline__ v16bf load_a_frag(const bf16_t* base, int ld) {
  const int lane = threadIdx.x & 31;
  const int r = lane & 15, g = lane >> 4;
  const bf16_t* p = base + (size_t)r * ld + g * 8;
  v8bf lo = *(const v8bf*)(p);
  v8bf hi = *(const v8bf*)(p + 16);
  return cat8(lo, hi);
}

// B-matrix fragment, 32x16 bf16, loaded from B^T stored row-major [N][K...] stride ld.
// ISA layout: lane holds column n=lane%16; K = g*8+{0..7} and 16+g*8+{0..7}, g=lane/16.
__device__ __forceinline__ v16bf load_b_frag(const bf16_t* base, int ld) {
  const int lane = threadIdx.x & 31;
  const int c = lane & 15, g = lane >> 4;
  const bf16_t* p = base + (size_t)c * ld + g * 8;
  v8bf lo = *(const v8bf*)(p);
  v8bf hi = *(const v8bf*)(p + 16);
  return cat8(lo, hi);
}

// ---------------------------------------------------------------------------
// Conversion / transpose kernels (bandwidth-trivial; everything lives in L2)
// ---------------------------------------------------------------------------
__global__ void k_conv_tokens(const float* __restrict__ x, bf16_t* __restrict__ y, int n) {
  int i = blockIdx.x * blockDim.x + threadIdx.x;
  if (i < n) y[i] = (bf16_t)x[i];
}

// Wq/Wk/Wv [H][C][D] f32 -> wt3 [3][H][D][C] bf16 (B^T layout for projections)
// grid (C*D/256, H, 3): pure shift/mask index math.
__global__ void k_conv_wqkv(const float* __restrict__ wq, const float* __restrict__ wk,
                            const float* __restrict__ wv, bf16_t* __restrict__ wt3) {
  int i = blockIdx.x * blockDim.x + threadIdx.x; // 0 .. C*D-1
  int h = blockIdx.y, z = blockIdx.z;
  int c = i >> 6, d = i & 63;
  const float* src = (z == 0) ? wq : (z == 1) ? wk : wv;
  wt3[(((size_t)z * H_HEADS + h) * D_HEAD + d) * C_DIM + c] =
      (bf16_t)src[((size_t)h * C_DIM + c) * D_HEAD + d];
}

// Wp [H*D][C] f32 -> wpt [C][H*D] bf16 (B^T for output projection)
__global__ void k_conv_wp(const float* __restrict__ wp, bf16_t* __restrict__ wpt) {
  int i = blockIdx.x * blockDim.x + threadIdx.x;
  if (i >= C_DIM * C_DIM) return;
  int n = i >> 10, k = i & 1023;
  wpt[i] = (bf16_t)wp[(size_t)k * C_DIM + n];
}

// ---------------------------------------------------------------------------
// QKV projections: per head [4096x1024]x[1024x64].
// grid (T/128, H, 3), 128 threads (4 waves); wave owns a 32(M) x 64(N) strip.
// Software-pipelined, unroll_count(2) => ping-pong buffers, no rotation movs.
// Q,K stored [H][T][D]; V stored transposed [H][D][T].
// ---------------------------------------------------------------------------
__global__ void __launch_bounds__(128, 1)
k_qkv_gemm(const bf16_t* __restrict__ xb, const bf16_t* __restrict__ wt3,
           bf16_t* __restrict__ qb, bf16_t* __restrict__ kb,
           bf16_t* __restrict__ vt) {
  const int wid = threadIdx.x >> 5, lane = threadIdx.x & 31;
  const int h = blockIdx.y, z = blockIdx.z;
  const int t0 = blockIdx.x * 128 + wid * 32;
  const bf16_t* wtb = wt3 + (((size_t)z * H_HEADS + h) * D_HEAD) * C_DIM; // [64][1024]
  const bf16_t* arow0 = xb + (size_t)t0 * C_DIM;
  const bf16_t* arow1 = arow0 + (size_t)16 * C_DIM;

  v8f acc[2][4] = {};

  // prologue: fragment buffers for kc = 0
  v16bf a0 = load_a_frag(arow0, C_DIM);
  v16bf a1 = load_a_frag(arow1, C_DIM);
  v16bf b0 = load_b_frag(wtb, C_DIM);
  v16bf b1 = load_b_frag(wtb + (size_t)16 * C_DIM, C_DIM);
  v16bf b2 = load_b_frag(wtb + (size_t)32 * C_DIM, C_DIM);
  v16bf b3 = load_b_frag(wtb + (size_t)48 * C_DIM, C_DIM);

#pragma clang loop unroll_count(2)
  for (int kc = 32; kc < C_DIM; kc += 32) {
    __builtin_prefetch(arow0 + kc + 128, 0, 1);
    // issue next iteration's loads first (overlap with the 8 WMMAs below)
    v16bf a0n = load_a_frag(arow0 + kc, C_DIM);
    v16bf a1n = load_a_frag(arow1 + kc, C_DIM);
    v16bf b0n = load_b_frag(wtb + kc, C_DIM);
    v16bf b1n = load_b_frag(wtb + (size_t)16 * C_DIM + kc, C_DIM);
    v16bf b2n = load_b_frag(wtb + (size_t)32 * C_DIM + kc, C_DIM);
    v16bf b3n = load_b_frag(wtb + (size_t)48 * C_DIM + kc, C_DIM);

    acc[0][0] = wmma_bf16(a0, b0, acc[0][0]);
    acc[1][0] = wmma_bf16(a1, b0, acc[1][0]);
    acc[0][1] = wmma_bf16(a0, b1, acc[0][1]);
    acc[1][1] = wmma_bf16(a1, b1, acc[1][1]);
    acc[0][2] = wmma_bf16(a0, b2, acc[0][2]);
    acc[1][2] = wmma_bf16(a1, b2, acc[1][2]);
    acc[0][3] = wmma_bf16(a0, b3, acc[0][3]);
    acc[1][3] = wmma_bf16(a1, b3, acc[1][3]);

    a0 = a0n; a1 = a1n; b0 = b0n; b1 = b1n; b2 = b2n; b3 = b3n;
  }
  // epilogue k-step
  acc[0][0] = wmma_bf16(a0, b0, acc[0][0]);
  acc[1][0] = wmma_bf16(a1, b0, acc[1][0]);
  acc[0][1] = wmma_bf16(a0, b1, acc[0][1]);
  acc[1][1] = wmma_bf16(a1, b1, acc[1][1]);
  acc[0][2] = wmma_bf16(a0, b2, acc[0][2]);
  acc[1][2] = wmma_bf16(a1, b2, acc[1][2]);
  acc[0][3] = wmma_bf16(a0, b3, acc[0][3]);
  acc[1][3] = wmma_bf16(a1, b3, acc[1][3]);

  const int nl = lane & 15, g = lane >> 4;
#pragma unroll
  for (int mt = 0; mt < 2; ++mt) {
#pragma unroll
    for (int n = 0; n < 4; ++n) {
#pragma unroll
      for (int j = 0; j < 8; ++j) {
        int t = t0 + mt * 16 + g * 8 + j;
        int d = n * 16 + nl;
        bf16_t v = (bf16_t)acc[mt][n][j];
        if (z == 0)      qb[((size_t)h * T_TOK + t) * D_HEAD + d] = v;
        else if (z == 1) kb[((size_t)h * T_TOK + t) * D_HEAD + d] = v;
        else             vt[((size_t)h * D_HEAD + d) * T_TOK + t] = v;
      }
    }
  }
}

// ---------------------------------------------------------------------------
// Flash attention (causal, online softmax). grid (T/64, H), 128 threads.
// Each wave owns one 16-row query tile; key blocks of 32.
// V frags + next key block's K frags are issued before the softmax VALU work
// so their latency hides behind exp/shuffle/LDS traffic.
// ---------------------------------------------------------------------------
__global__ void __launch_bounds__(128, 1)
k_flash(const bf16_t* __restrict__ qb, const bf16_t* __restrict__ kb,
        const bf16_t* __restrict__ vt, bf16_t* __restrict__ att) {
  __shared__ __align__(16) bf16_t pstage[4][16 * 32];
  const int wid = threadIdx.x >> 5, lane = threadIdx.x & 31;
  const int nl = lane & 15, g = lane >> 4;
  const int h = blockIdx.y;
  const int t0 = blockIdx.x * 64 + wid * 16;
  const bf16_t* qh = qb + (size_t)h * T_TOK * D_HEAD;
  const bf16_t* kh = kb + (size_t)h * T_TOK * D_HEAD;
  const bf16_t* vh = vt + (size_t)h * D_HEAD * T_TOK;
  bf16_t* ps = pstage[wid];

  // Q tile fragments: K-dim = D = 64 -> two 16x32 A-frags, loaded once.
  v16bf a0 = load_a_frag(qh + (size_t)t0 * D_HEAD, D_HEAD);
  v16bf a1 = load_a_frag(qh + (size_t)t0 * D_HEAD + 32, D_HEAD);

  v8f o[4] = {};
  float m[8], l[8];
#pragma unroll
  for (int j = 0; j < 8; ++j) { m[j] = -3.0e38f; l[j] = 0.f; }

  const float scale = 0.125f; // 1/sqrt(64)
  const int s_end = t0 + 16;

  // preload key block 0 fragments (double-buffered across iterations)
  v16bf bk0 = load_b_frag(kh, D_HEAD);
  v16bf bk1 = load_b_frag(kh + 32, D_HEAD);
  v16bf bk2 = load_b_frag(kh + (size_t)16 * D_HEAD, D_HEAD);
  v16bf bk3 = load_b_frag(kh + (size_t)16 * D_HEAD + 32, D_HEAD);

#pragma clang loop unroll_count(2)
  for (int s0 = 0; s0 < s_end; s0 += 32) {
    // scores: two 16x16 tiles (cols s0..s0+15, s0+16..s0+31)
    v8f sa = {}, sb = {};
    sa = wmma_bf16(a0, bk0, sa);
    sa = wmma_bf16(a1, bk1, sa);
    sb = wmma_bf16(a0, bk2, sb);
    sb = wmma_bf16(a1, bk3, sb);

    // issue V-fragment loads for this block early
    v16bf bv0 = load_b_frag(vh + (size_t)0 * T_TOK + s0, T_TOK);
    v16bf bv1 = load_b_frag(vh + (size_t)16 * T_TOK + s0, T_TOK);
    v16bf bv2 = load_b_frag(vh + (size_t)32 * T_TOK + s0, T_TOK);
    v16bf bv3 = load_b_frag(vh + (size_t)48 * T_TOK + s0, T_TOK);

    // issue next key block's K-fragment loads early (clamped on last iter)
    const int s1 = (s0 + 32 < s_end) ? s0 + 32 : s0;
    const bf16_t* knx = kh + (size_t)s1 * D_HEAD;
    bk0 = load_b_frag(knx, D_HEAD);
    bk1 = load_b_frag(knx + 32, D_HEAD);
    bk2 = load_b_frag(knx + (size_t)16 * D_HEAD, D_HEAD);
    bk3 = load_b_frag(knx + (size_t)16 * D_HEAD + 32, D_HEAD);

    // online softmax (VALU + cross-lane + LDS staging hides load latency)
#pragma unroll
    for (int j = 0; j < 8; ++j) {
      const int trow = t0 + g * 8 + j;
      float v0 = (s0 + nl      <= trow) ? sa[j] * scale : -3.0e38f;
      float v1 = (s0 + 16 + nl <= trow) ? sb[j] * scale : -3.0e38f;
      float mb = fmaxf(v0, v1);
#pragma unroll
      for (int msk = 8; msk >= 1; msk >>= 1) mb = fmaxf(mb, __shfl_xor(mb, msk, 32));
      float mn = fmaxf(m[j], mb);
      float p0 = __expf(v0 - mn), p1 = __expf(v1 - mn);
      float corr = __expf(m[j] - mn);
      float rs = p0 + p1;
#pragma unroll
      for (int msk = 8; msk >= 1; msk >>= 1) rs += __shfl_xor(rs, msk, 32);
      l[j] = l[j] * corr + rs;
      m[j] = mn;
#pragma unroll
      for (int n = 0; n < 4; ++n) o[n][j] *= corr;
      // stage P (C-layout -> LDS row-major [16][32] for A-layout reload)
      ps[(g * 8 + j) * 32 + nl]      = (bf16_t)p0;
      ps[(g * 8 + j) * 32 + 16 + nl] = (bf16_t)p1;
    }

    v16bf pf = load_a_frag(ps, 32);
    o[0] = wmma_bf16(pf, bv0, o[0]);
    o[1] = wmma_bf16(pf, bv1, o[1]);
    o[2] = wmma_bf16(pf, bv2, o[2]);
    o[3] = wmma_bf16(pf, bv3, o[3]);
  }

  // normalize + store into concat buffer att[T][H*D] (head-major columns)
#pragma unroll
  for (int j = 0; j < 8; ++j) {
    float inv = 1.0f / l[j];
    int t = t0 + g * 8 + j;
#pragma unroll
    for (int n = 0; n < 4; ++n) {
      int col = h * D_HEAD + n * 16 + nl;
      att[(size_t)t * C_DIM + col] = (bf16_t)(o[n][j] * inv);
    }
  }
}

// ---------------------------------------------------------------------------
// Output projection: [4096x1024] x [1024x1024] + bias -> f32 out.
// grid (T/128, C/64), 128 threads; wave owns 32(M) x 64(N). Software-pipelined.
// ---------------------------------------------------------------------------
__global__ void __launch_bounds__(128, 1)
k_out_gemm(const bf16_t* __restrict__ att, const bf16_t* __restrict__ wpt,
           const float* __restrict__ bias, float* __restrict__ out) {
  const int wid = threadIdx.x >> 5, lane = threadIdx.x & 31;
  const int t0 = blockIdx.x * 128 + wid * 32;
  const int n0 = blockIdx.y * 64;
  const bf16_t* arow0 = att + (size_t)t0 * C_DIM;
  const bf16_t* arow1 = arow0 + (size_t)16 * C_DIM;
  const bf16_t* wcol = wpt + (size_t)n0 * C_DIM;

  v8f acc[2][4] = {};

  v16bf a0 = load_a_frag(arow0, C_DIM);
  v16bf a1 = load_a_frag(arow1, C_DIM);
  v16bf b0 = load_b_frag(wcol, C_DIM);
  v16bf b1 = load_b_frag(wcol + (size_t)16 * C_DIM, C_DIM);
  v16bf b2 = load_b_frag(wcol + (size_t)32 * C_DIM, C_DIM);
  v16bf b3 = load_b_frag(wcol + (size_t)48 * C_DIM, C_DIM);

#pragma clang loop unroll_count(2)
  for (int kc = 32; kc < C_DIM; kc += 32) {
    __builtin_prefetch(arow0 + kc + 128, 0, 1);
    v16bf a0n = load_a_frag(arow0 + kc, C_DIM);
    v16bf a1n = load_a_frag(arow1 + kc, C_DIM);
    v16bf b0n = load_b_frag(wcol + kc, C_DIM);
    v16bf b1n = load_b_frag(wcol + (size_t)16 * C_DIM + kc, C_DIM);
    v16bf b2n = load_b_frag(wcol + (size_t)32 * C_DIM + kc, C_DIM);
    v16bf b3n = load_b_frag(wcol + (size_t)48 * C_DIM + kc, C_DIM);

    acc[0][0] = wmma_bf16(a0, b0, acc[0][0]);
    acc[1][0] = wmma_bf16(a1, b0, acc[1][0]);
    acc[0][1] = wmma_bf16(a0, b1, acc[0][1]);
    acc[1][1] = wmma_bf16(a1, b1, acc[1][1]);
    acc[0][2] = wmma_bf16(a0, b2, acc[0][2]);
    acc[1][2] = wmma_bf16(a1, b2, acc[1][2]);
    acc[0][3] = wmma_bf16(a0, b3, acc[0][3]);
    acc[1][3] = wmma_bf16(a1, b3, acc[1][3]);

    a0 = a0n; a1 = a1n; b0 = b0n; b1 = b1n; b2 = b2n; b3 = b3n;
  }
  acc[0][0] = wmma_bf16(a0, b0, acc[0][0]);
  acc[1][0] = wmma_bf16(a1, b0, acc[1][0]);
  acc[0][1] = wmma_bf16(a0, b1, acc[0][1]);
  acc[1][1] = wmma_bf16(a1, b1, acc[1][1]);
  acc[0][2] = wmma_bf16(a0, b2, acc[0][2]);
  acc[1][2] = wmma_bf16(a1, b2, acc[1][2]);
  acc[0][3] = wmma_bf16(a0, b3, acc[0][3]);
  acc[1][3] = wmma_bf16(a1, b3, acc[1][3]);

  const int nl = lane & 15, g = lane >> 4;
#pragma unroll
  for (int mt = 0; mt < 2; ++mt) {
#pragma unroll
    for (int n = 0; n < 4; ++n) {
      int col = n0 + n * 16 + nl;
      float bv = bias[col];
#pragma unroll
      for (int j = 0; j < 8; ++j) {
        int t = t0 + mt * 16 + g * 8 + j;
        out[(size_t)t * C_DIM + col] = acc[mt][n][j] + bv;
      }
    }
  }
}

// ---------------------------------------------------------------------------
extern "C" void kernel_launch(void* const* d_in, const int* in_sizes, int n_in,
                              void* d_out, int out_size, void* d_ws, size_t ws_size,
                              hipStream_t stream) {
  (void)in_sizes; (void)n_in; (void)out_size; (void)ws_size;
  const float* tokens = (const float*)d_in[0];
  const float* wq = (const float*)d_in[1];
  const float* wk = (const float*)d_in[2];
  const float* wv = (const float*)d_in[3];
  const float* wp = (const float*)d_in[4];
  const float* bp = (const float*)d_in[5];
  float* out = (float*)d_out;

  char* ws = (char*)d_ws;
  size_t off = 0;
  bf16_t* xb  = (bf16_t*)(ws + off); off += (size_t)T_TOK * C_DIM * 2;                 // 8 MB
  bf16_t* wt3 = (bf16_t*)(ws + off); off += (size_t)3 * H_HEADS * D_HEAD * C_DIM * 2;  // 6 MB
  bf16_t* wpt = (bf16_t*)(ws + off); off += (size_t)C_DIM * C_DIM * 2;                 // 2 MB
  bf16_t* qb  = (bf16_t*)(ws + off); off += (size_t)H_HEADS * T_TOK * D_HEAD * 2;      // 8 MB
  bf16_t* kb  = (bf16_t*)(ws + off); off += (size_t)H_HEADS * T_TOK * D_HEAD * 2;      // 8 MB
  bf16_t* vt  = (bf16_t*)(ws + off); off += (size_t)H_HEADS * D_HEAD * T_TOK * 2;      // 8 MB
  bf16_t* att = (bf16_t*)(ws + off); off += (size_t)T_TOK * C_DIM * 2;                 // 8 MB

  const int ntok = T_TOK * C_DIM;
  k_conv_tokens<<<(ntok + 255) / 256, 256, 0, stream>>>(tokens, xb, ntok);
  k_conv_wqkv<<<dim3((C_DIM * D_HEAD) / 256, H_HEADS, 3), 256, 0, stream>>>(wq, wk, wv, wt3);
  k_conv_wp<<<(C_DIM * C_DIM + 255) / 256, 256, 0, stream>>>(wp, wpt);

  k_qkv_gemm<<<dim3(T_TOK / 128, H_HEADS, 3), 128, 0, stream>>>(xb, wt3, qb, kb, vt);
  k_flash<<<dim3(T_TOK / 64, H_HEADS), 128, 0, stream>>>(qb, kb, vt, att);
  k_out_gemm<<<dim3(T_TOK / 128, C_DIM / 64), 128, 0, stream>>>(att, wpt, bp, out);
}
